// FCOSModule_56238301774055
// MI455X (gfx1250) — matrix operand order repaired and checked by reading, not compiled, not run on hardware
//
#include <hip/hip_runtime.h>
#include <hip/hip_bf16.h>
#include <math.h>

// ---------------------------------------------------------------------------
// FCOS head on MI455X (gfx1250, wave32).
// conv1d(K=3) = 3 shifted GEMMs accumulated in v_wmma_f32_16x16x32_bf16.
// Activations: padded [B][T+2][C] channel-contiguous bf16 (zero pad rows), so
// shifts are row offsets and all vector memory ops are aligned b128.
// Per K-chunk the block cooperatively stages weights (64x32 per tap) and a
// 130x32 activation slab into LDS; all 3 taps reuse one staged slab. WMMA
// fragments are then read with ds_read_b128 in the documented register
// layouts. Weights are prefetched one chunk ahead (global_prefetch_b8).
// ---------------------------------------------------------------------------

typedef __bf16 bf16;
typedef __attribute__((ext_vector_type(16))) __bf16 bf16x16;
typedef __attribute__((ext_vector_type(8)))  float  floatx8;

union Frag { bf16x16 v; uint4 q[2]; };

static inline int cdiv_i(int a, int b) { return (a + b - 1) / b; }

#define B_BATCH 8
#define CBUF 512            // channel stride of all bf16 activation buffers (except h)

// ---------------------------------------------------------------------------
// Core WMMA conv kernel: out[b, t, co] = relu( scale[co] * conv + shift[co] )
//   input:  srcA/srcB  [B][T+2][512] bf16 (concat along channels at splitC)
//   weight: W          [KR][Cout][CinTot] bf16
//   output: out        [B][T+2][Cout] bf16 (writes rows 1..T; pads zeroed elsewhere)
// Block: 256 threads = 8 waves; block tile 64(M=co) x 128(N=t); wave tile 32x32.
// ---------------------------------------------------------------------------
template <int KR>
__global__ __launch_bounds__(256)
void k_conv_wmma(const bf16* __restrict__ srcA, const bf16* __restrict__ srcB,
                 int splitC, const bf16* __restrict__ W, int CinTot, int Cout,
                 const float* __restrict__ sc, const float* __restrict__ sh,
                 bf16* __restrict__ out, int T) {
  // LDS staging: A = [KR][64 co][32 ci], B = [130 t-rows][32 ci]
  __shared__ alignas(16) bf16 sA[KR * 64 * 32];
  __shared__ alignas(16) bf16 sB[130 * 32];

  const int Tp   = T + 2;
  const int b    = blockIdx.z;
  const int n0   = blockIdx.x * 128;          // t tile base
  const int m0   = blockIdx.y * 64;           // co tile base
  const int tid  = threadIdx.x;
  const int wave = tid >> 5;
  const int lane = tid & 31;
  const int lr   = lane & 15;
  const int hi   = lane >> 4;                 // 0/1 half of wave
  const int mL   = (wave >> 2) * 32;          // wave M base within block (2 waves on M)
  const int nL   = (wave & 3) * 32;           // wave N base within block (4 waves on N)
  const int rowBase = (KR == 3) ? 0 : 1;      // padded-row offset so taps span t-1..t+1

  floatx8 acc[2][2];
#pragma unroll
  for (int i = 0; i < 2; ++i)
#pragma unroll
    for (int j = 0; j < 2; ++j) acc[i][j] = (floatx8)0.0f;

  for (int ci0 = 0; ci0 < CinTot; ci0 += 32) {
    const bf16* xb; int cic;
    if (ci0 < splitC) { xb = srcA; cic = ci0; }
    else              { xb = srcB; cic = ci0 - splitC; }

    __syncthreads();  // protect previous iteration's LDS reads

    // ---- stage B slab: padded rows n0 .. n0+129, 32 channels (520 uint4)
    for (int i = tid; i < 130 * 4; i += 256) {
      const int row = i >> 2, q = i & 3;
      *(uint4*)&sB[row * 32 + q * 8] =
          *(const uint4*)(xb + ((size_t)b * Tp + n0 + row) * CBUF + cic + q * 8);
    }
    // ---- stage A slab: KR taps x 64 co rows x 32 ci (KR*256 uint4)
    for (int i = tid; i < KR * 256; i += 256) {
      const int k = i >> 8, r = (i >> 2) & 63, q = i & 3;
      *(uint4*)&sA[(k * 64 + r) * 32 + q * 8] =
          *(const uint4*)(W + ((size_t)k * Cout + m0 + r) * CinTot + ci0 + q * 8);
    }
    // prefetch next chunk of the weight stream into cache
    if (ci0 + 32 < CinTot)
      __builtin_prefetch(W + ((size_t)m0) * CinTot + ci0 + 32, 0, 1);

    __syncthreads();

#pragma unroll
    for (int k = 0; k < KR; ++k) {
      // ---- B fragments (32x16, K=ci): lane<16 K={0..7,16..23}, lane>=16 K={8..15,24..31}
      Frag bf[2];
#pragma unroll
      for (int ns = 0; ns < 2; ++ns) {
        const int tl = nL + ns * 16 + lr + k + rowBase;   // local padded row
        bf[ns].q[0] = *(const uint4*)&sB[tl * 32 + hi * 8];
        bf[ns].q[1] = *(const uint4*)&sB[tl * 32 + hi * 8 + 16];
      }
      // ---- A fragments (16x32): lane<16 K={0..7,16..23}, lane>=16 K={8..15,24..31}
      Frag af[2];
#pragma unroll
      for (int ms = 0; ms < 2; ++ms) {
        const int rl = mL + ms * 16 + lr;                 // local co row
        af[ms].q[0] = *(const uint4*)&sA[(k * 64 + rl) * 32 + hi * 8];
        af[ms].q[1] = *(const uint4*)&sA[(k * 64 + rl) * 32 + hi * 8 + 16];
      }
#pragma unroll
      for (int ms = 0; ms < 2; ++ms)
#pragma unroll
        for (int ns = 0; ns < 2; ++ns)
          acc[ms][ns] = __builtin_amdgcn_wmma_f32_16x16x32_bf16(
              false, af[ms].v, false, bf[ns].v, (short)0, acc[ms][ns], false, false);
    }
  }

  // ---- Epilogue: BN(scale,shift) + ReLU, bf16 pack. C/D layout: VGPR r holds
  // co = base + hi*8 + r (8 contiguous channels per lane) -> one b128 store.
#pragma unroll
  for (int ms = 0; ms < 2; ++ms) {
    const int coB = m0 + mL + ms * 16 + hi * 8;
    const float4 s0 = *(const float4*)(sc + coB);
    const float4 s1 = *(const float4*)(sc + coB + 4);
    const float4 h0 = *(const float4*)(sh + coB);
    const float4 h1 = *(const float4*)(sh + coB + 4);
    const float ss[8] = {s0.x, s0.y, s0.z, s0.w, s1.x, s1.y, s1.z, s1.w};
    const float hh[8] = {h0.x, h0.y, h0.z, h0.w, h1.x, h1.y, h1.z, h1.w};
#pragma unroll
    for (int ns = 0; ns < 2; ++ns) {
      const int t = n0 + nL + ns * 16 + lr;
      union { uint4 q; bf16 e[8]; } o;
#pragma unroll
      for (int r = 0; r < 8; ++r) {
        float y = acc[ms][ns][r] * ss[r] + hh[r];
        y = y > 0.0f ? y : 0.0f;
        o.e[r] = (bf16)y;
      }
      *(uint4*)(out + (((size_t)b * Tp + (t + 1)) * Cout + coB)) = o.q;
    }
  }
}

// ---------------------------------------------------------------------------
// Prep kernels (run every call; deterministic, tiny vs. main GEMMs)
// ---------------------------------------------------------------------------

// f [B][512][T] fp32 -> xb [B][T+2][512] bf16 with zeroed pad rows 0 and T+1
__global__ void k_cvt_in(const float* __restrict__ f, bf16* __restrict__ xb, int T) {
  const int Tp = T + 2;
  const long long n = (long long)B_BATCH * Tp * CBUF;
  long long idx = (long long)blockIdx.x * blockDim.x + threadIdx.x;
  if (idx >= n) return;
  int ci = (int)(idx % CBUF);
  long long r = idx / CBUF;
  int trow = (int)(r % Tp);
  int b = (int)(r / Tp);
  float v = 0.0f;
  if (trow >= 1 && trow <= T) v = f[((size_t)b * CBUF + ci) * T + (trow - 1)];
  xb[idx] = (bf16)v;
}

// tower weights [L][512][512][3] fp32 -> [L][3][512][512] bf16
__global__ void k_cvt_tw(const float* __restrict__ w, bf16* __restrict__ o, int n) {
  int idx = blockIdx.x * blockDim.x + threadIdx.x;
  if (idx >= n) return;
  int ci = idx % 512;
  int t1 = idx / 512;
  int co = t1 % 512;
  int t2 = t1 / 512;
  int k  = t2 % 3;
  int l  = t2 / 3;
  o[idx] = (bf16)w[(((size_t)l * 512 + co) * 512 + ci) * 3 + k];
}

// plain fp32 -> bf16 (mix weight is already [co][ci] with K=1)
__global__ void k_cvt_f2b(const float* __restrict__ w, bf16* __restrict__ o, int n) {
  int idx = blockIdx.x * blockDim.x + threadIdx.x;
  if (idx >= n) return;
  o[idx] = (bf16)w[idx];
}

// iou conv1 weights [256][512][3] fp32 -> [3][256][512] bf16
__global__ void k_cvt_iou1(const float* __restrict__ w, bf16* __restrict__ o) {
  int idx = blockIdx.x * blockDim.x + threadIdx.x;
  const int n = 3 * 256 * 512;
  if (idx >= n) return;
  int ci = idx % 512;
  int co = (idx / 512) % 256;
  int k  = idx / (512 * 256);
  o[idx] = (bf16)w[((size_t)co * 512 + ci) * 3 + k];
}

// fold conv bias + BN into per-channel scale/shift:  y = acc*s + t
__global__ void k_bnprep(const float* __restrict__ g, const float* __restrict__ be,
                         const float* __restrict__ m, const float* __restrict__ v,
                         const float* __restrict__ cb, float* __restrict__ sc,
                         float* __restrict__ sh, int n) {
  int i = blockIdx.x * blockDim.x + threadIdx.x;
  if (i >= n) return;
  float s = g[i] * rsqrtf(v[i] + 1e-5f);
  sc[i] = s;
  sh[i] = (cb[i] - m[i]) * s + be[i];
}

// zero pad rows 0 and T+1 of an activation buffer
__global__ void k_zpad(bf16* __restrict__ p, int T, int Cb) {
  int idx = blockIdx.x * blockDim.x + threadIdx.x;
  const int n = B_BATCH * 2 * Cb;
  if (idx >= n) return;
  int ci = idx % Cb;
  int wch = (idx / Cb) & 1;
  int b = idx / (2 * Cb);
  int row = wch ? (T + 1) : 0;
  p[((size_t)b * (T + 2) + row) * Cb + ci] = (bf16)0.0f;
}

// ---------------------------------------------------------------------------
// Small heads: logits(ct), bbox=exp(scale*conv(bt)), iou2(h). One thread per (b,t).
// ---------------------------------------------------------------------------
__global__ void k_heads(const bf16* __restrict__ ct, const bf16* __restrict__ bt,
                        const bf16* __restrict__ h,
                        const float* __restrict__ lw, const float* __restrict__ lb,
                        const float* __restrict__ bw, const float* __restrict__ bb,
                        const float* __restrict__ iw2, const float* __restrict__ ib2,
                        const float* __restrict__ scales, int lvl, int T, int toff,
                        float* __restrict__ out) {
  int idx = blockIdx.x * blockDim.x + threadIdx.x;
  if (idx >= B_BATCH * T) return;
  const int b = idx / T;
  const int t = idx % T;
  const int Tp = T + 2;

  float lg  = lb[0];
  float bx0 = bb[0];
  float bx1 = bb[1];
  const bf16* ctb = ct + (size_t)b * Tp * CBUF;
  const bf16* btb = bt + (size_t)b * Tp * CBUF;
  for (int k = 0; k < 3; ++k) {
    const bf16* crow = ctb + (size_t)(t + k) * CBUF;  // padded: covers t-1..t+1
    const bf16* brow = btb + (size_t)(t + k) * CBUF;
    for (int ci = 0; ci < CBUF; ++ci) {
      float cv = (float)crow[ci];
      float bv = (float)brow[ci];
      lg  += cv * lw[ci * 3 + k];
      bx0 += bv * bw[(0 * CBUF + ci) * 3 + k];
      bx1 += bv * bw[(1 * CBUF + ci) * 3 + k];
    }
  }
  float iou = ib2[0];
  const bf16* hrow = h + ((size_t)b * Tp + (t + 1)) * 256;
  for (int ci = 0; ci < 256; ++ci) iou += (float)hrow[ci] * iw2[ci];

  const float s = scales[lvl];
  const int TT = 1792;
  out[((size_t)b * 4 + 0) * TT + toff + t] = lg;
  out[((size_t)b * 4 + 1) * TT + toff + t] = expf(bx0 * s);
  out[((size_t)b * 4 + 2) * TT + toff + t] = expf(bx1 * s);
  out[((size_t)b * 4 + 3) * TT + toff + t] = iou;
}

// ---------------------------------------------------------------------------
// Launch
// ---------------------------------------------------------------------------
extern "C" void kernel_launch(void* const* d_in, const int* in_sizes, int n_in,
                              void* d_out, int out_size, void* d_ws, size_t ws_size,
                              hipStream_t stream) {
  (void)in_sizes; (void)n_in; (void)out_size; (void)ws_size;

  const float* f[3]     = {(const float*)d_in[0], (const float*)d_in[1], (const float*)d_in[2]};
  const float* cls_w    = (const float*)d_in[3];
  const float* cls_b    = (const float*)d_in[4];
  const float* cls_g    = (const float*)d_in[5];
  const float* cls_be   = (const float*)d_in[6];
  const float* cls_m    = (const float*)d_in[7];
  const float* cls_v    = (const float*)d_in[8];
  const float* box_w    = (const float*)d_in[9];
  const float* box_b    = (const float*)d_in[10];
  const float* box_g    = (const float*)d_in[11];
  const float* box_be   = (const float*)d_in[12];
  const float* box_m    = (const float*)d_in[13];
  const float* box_v    = (const float*)d_in[14];
  const float* logits_w = (const float*)d_in[15];
  const float* logits_b = (const float*)d_in[16];
  const float* bbox_w   = (const float*)d_in[17];
  const float* bbox_b   = (const float*)d_in[18];
  const float* mix_w    = (const float*)d_in[19];
  const float* mix_b    = (const float*)d_in[20];
  const float* mix_g    = (const float*)d_in[21];
  const float* mix_be   = (const float*)d_in[22];
  const float* mix_m    = (const float*)d_in[23];
  const float* mix_v    = (const float*)d_in[24];
  const float* iou_w1   = (const float*)d_in[25];
  const float* iou_b1   = (const float*)d_in[26];
  const float* iou_g    = (const float*)d_in[27];
  const float* iou_be   = (const float*)d_in[28];
  const float* iou_m    = (const float*)d_in[29];
  const float* iou_v    = (const float*)d_in[30];
  const float* iou_w2   = (const float*)d_in[31];
  const float* iou_b2   = (const float*)d_in[32];
  const float* scales   = (const float*)d_in[33];
  float* out = (float*)d_out;

  // ---- workspace layout -------------------------------------------------
  char* ws = (char*)d_ws;
  size_t off = 0;
  auto take = [&](size_t bytes) -> char* {
    char* p = ws + off;
    off = (off + bytes + 255) & ~(size_t)255;
    return p;
  };
  const size_t ABYTES  = (size_t)B_BATCH * 1026 * CBUF * sizeof(bf16);  // max-T act buf
  bf16* xb   = (bf16*)take(ABYTES);
  bf16* buf1 = (bf16*)take(ABYTES);
  bf16* buf2 = (bf16*)take(ABYTES);
  bf16* ctB  = (bf16*)take(ABYTES);
  bf16* btB  = (bf16*)take(ABYTES);
  bf16* mixB = (bf16*)take(ABYTES);
  bf16* hB   = (bf16*)take((size_t)B_BATCH * 1026 * 256 * sizeof(bf16));
  bf16* Wcls = (bf16*)take((size_t)4 * 3 * 512 * 512 * sizeof(bf16));
  bf16* Wbox = (bf16*)take((size_t)4 * 3 * 512 * 512 * sizeof(bf16));
  bf16* Wmix = (bf16*)take((size_t)512 * 1024 * sizeof(bf16));
  bf16* Wio1 = (bf16*)take((size_t)3 * 256 * 512 * sizeof(bf16));
  float* scCls = (float*)take(4 * 512 * sizeof(float));
  float* shCls = (float*)take(4 * 512 * sizeof(float));
  float* scBox = (float*)take(4 * 512 * sizeof(float));
  float* shBox = (float*)take(4 * 512 * sizeof(float));
  float* scMix = (float*)take(512 * sizeof(float));
  float* shMix = (float*)take(512 * sizeof(float));
  float* scIou = (float*)take(256 * sizeof(float));
  float* shIou = (float*)take(256 * sizeof(float));

  const int BT = 256;  // block threads

  // ---- one-time per-call prep (weights -> bf16 layouts, BN folding) -----
  {
    int n = 4 * 3 * 512 * 512;
    k_cvt_tw<<<cdiv_i(n, BT), BT, 0, stream>>>(cls_w, Wcls, n);
    k_cvt_tw<<<cdiv_i(n, BT), BT, 0, stream>>>(box_w, Wbox, n);
    int nm = 512 * 1024;
    k_cvt_f2b<<<cdiv_i(nm, BT), BT, 0, stream>>>(mix_w, Wmix, nm);
    k_cvt_iou1<<<cdiv_i(3 * 256 * 512, BT), BT, 0, stream>>>(iou_w1, Wio1);
    k_bnprep<<<cdiv_i(4 * 512, BT), BT, 0, stream>>>(cls_g, cls_be, cls_m, cls_v, cls_b,
                                                     scCls, shCls, 4 * 512);
    k_bnprep<<<cdiv_i(4 * 512, BT), BT, 0, stream>>>(box_g, box_be, box_m, box_v, box_b,
                                                     scBox, shBox, 4 * 512);
    k_bnprep<<<cdiv_i(512, BT), BT, 0, stream>>>(mix_g, mix_be, mix_m, mix_v, mix_b,
                                                 scMix, shMix, 512);
    k_bnprep<<<cdiv_i(256, BT), BT, 0, stream>>>(iou_g, iou_be, iou_m, iou_v, iou_b1,
                                                 scIou, shIou, 256);
  }

  const int Ts[3]   = {1024, 512, 256};
  const int toff[3] = {0, 1024, 1536};

  for (int l = 0; l < 3; ++l) {
    const int T = Ts[l];

    // input fp32 [B][C][T] -> padded bf16 [B][T+2][C]
    {
      long long n = (long long)B_BATCH * (T + 2) * CBUF;
      k_cvt_in<<<(unsigned)((n + BT - 1) / BT), BT, 0, stream>>>(f[l], xb, T);
    }
    // zero pad rows of every intermediate buffer at this level's T
    {
      int n = B_BATCH * 2 * CBUF;
      k_zpad<<<cdiv_i(n, BT), BT, 0, stream>>>(buf1, T, CBUF);
      k_zpad<<<cdiv_i(n, BT), BT, 0, stream>>>(buf2, T, CBUF);
      k_zpad<<<cdiv_i(n, BT), BT, 0, stream>>>(ctB, T, CBUF);
      k_zpad<<<cdiv_i(n, BT), BT, 0, stream>>>(btB, T, CBUF);
      k_zpad<<<cdiv_i(n, BT), BT, 0, stream>>>(mixB, T, CBUF);
    }

    dim3 g512(T / 128, 512 / 64, B_BATCH);  // Cout=512 convs
    dim3 g256(T / 128, 256 / 64, B_BATCH);  // iou conv1 (Cout=256)
    const size_t WL = (size_t)3 * 512 * 512;

    // cls tower: xb -> buf1 -> buf2 -> buf1 -> ctB
    k_conv_wmma<3><<<g512, BT, 0, stream>>>(xb,   xb,   512, Wcls + 0 * WL, 512, 512,
                                            scCls + 0 * 512, shCls + 0 * 512, buf1, T);
    k_conv_wmma<3><<<g512, BT, 0, stream>>>(buf1, buf1, 512, Wcls + 1 * WL, 512, 512,
                                            scCls + 1 * 512, shCls + 1 * 512, buf2, T);
    k_conv_wmma<3><<<g512, BT, 0, stream>>>(buf2, buf2, 512, Wcls + 2 * WL, 512, 512,
                                            scCls + 2 * 512, shCls + 2 * 512, buf1, T);
    k_conv_wmma<3><<<g512, BT, 0, stream>>>(buf1, buf1, 512, Wcls + 3 * WL, 512, 512,
                                            scCls + 3 * 512, shCls + 3 * 512, ctB, T);
    // box tower: xb -> buf1 -> buf2 -> buf1 -> btB
    k_conv_wmma<3><<<g512, BT, 0, stream>>>(xb,   xb,   512, Wbox + 0 * WL, 512, 512,
                                            scBox + 0 * 512, shBox + 0 * 512, buf1, T);
    k_conv_wmma<3><<<g512, BT, 0, stream>>>(buf1, buf1, 512, Wbox + 1 * WL, 512, 512,
                                            scBox + 1 * 512, shBox + 1 * 512, buf2, T);
    k_conv_wmma<3><<<g512, BT, 0, stream>>>(buf2, buf2, 512, Wbox + 2 * WL, 512, 512,
                                            scBox + 2 * 512, shBox + 2 * 512, buf1, T);
    k_conv_wmma<3><<<g512, BT, 0, stream>>>(buf1, buf1, 512, Wbox + 3 * WL, 512, 512,
                                            scBox + 3 * 512, shBox + 3 * 512, btB, T);
    // mix: 1x1 over concat[ct, bt] (CinTot=1024, split at 512)
    k_conv_wmma<1><<<g512, BT, 0, stream>>>(ctB, btB, 512, Wmix, 1024, 512,
                                            scMix, shMix, mixB, T);
    // iou conv1: K=3, 512 -> 256
    k_conv_wmma<3><<<g256, BT, 0, stream>>>(mixB, mixB, 1024, Wio1, 512, 256,
                                            scIou, shIou, hB, T);
    // small heads -> d_out [8][4][1792]
    k_heads<<<cdiv_i(B_BATCH * T, BT), BT, 0, stream>>>(
        ctB, btB, hB, logits_w, logits_b, bbox_w, bbox_b, iou_w2, iou_b2,
        scales, l, T, toff[l], out);
  }
}
